// TwistorSelfAttention_29403346108745
// MI455X (gfx1250) — compile-verified, matrix operand order checked by reading, not run on hardware
//
#include <hip/hip_runtime.h>
#include <hip/hip_bf16.h>
#include <stdint.h>

// ---------------------------------------------------------------------------
// Types for CDNA5 WMMA (wave32): A/B = v16h (16 halves = 8 VGPRs),
// C/D = v8f (8 VGPRs of f32).
// ---------------------------------------------------------------------------
typedef __attribute__((ext_vector_type(16))) _Float16 v16h;
typedef __attribute__((ext_vector_type(8)))  float    v8f;
typedef __attribute__((ext_vector_type(8)))  _Float16 f16x8;
typedef __attribute__((ext_vector_type(4)))  int      v4i;

#define SEQ      2048
#define HIDDEN   2048
#define NHEADS   32
#define NKV      8
#define HDIM     64
#define KVDIM    (NKV * HDIM)   // 512

// Async Global->LDS (ASYNCcnt-tracked) if the toolchain exposes it.
#if defined(__gfx1250__) && \
    __has_builtin(__builtin_amdgcn_global_load_async_to_lds_b128) && \
    __has_builtin(__builtin_amdgcn_s_wait_asynccnt)
#define HAVE_ASYNC 1
#else
#define HAVE_ASYNC 0
#endif

#if HAVE_ASYNC
// Builtin signature (from compiler diagnostic): typed v4i pointers,
// (addrspace(1) src, addrspace(3) dst, imm offset, imm cpol).
typedef __attribute__((address_space(1))) v4i GlobalV4;
typedef __attribute__((address_space(3))) v4i LocalV4;
#endif

// Copy 16 bytes Global -> LDS. Async path bypasses VGPRs (tracked by
// ASYNCcnt); fallback is a synchronous load+store.
__device__ __forceinline__ void cp16(const _Float16* g, _Float16* l) {
#if HAVE_ASYNC
  __builtin_amdgcn_global_load_async_to_lds_b128(
      (GlobalV4*)(const void*)g, (LocalV4*)(void*)l, 0, 0);
#else
  *(f16x8*)l = *(const f16x8*)g;
#endif
}
__device__ __forceinline__ void async_wait_le4() {
#if HAVE_ASYNC
  __builtin_amdgcn_s_wait_asynccnt(4);   // keep newest 4 (next tile) in flight
#endif
}
__device__ __forceinline__ void async_wait_0() {
#if HAVE_ASYNC
  __builtin_amdgcn_s_wait_asynccnt(0);
#endif
}

__device__ __forceinline__ v8f wmma16(v16h a, v16h b, v8f c) {
  // D = A(16x32 f16) * B(32x16 f16) + C(16x16 f32)
  return __builtin_amdgcn_wmma_f32_16x16x32_f16(
      /*neg_a=*/false, a, /*neg_b=*/false, b,
      /*c_mod=*/(short)0, c, /*reuse_a=*/false, /*reuse_b=*/false);
}

// One lane's 16-element WMMA fragment: two contiguous 8-element (16 B) runs
// 16 elements apart in K (e0..7 -> K = 8*half + e ; e8..15 -> K = 16+8*half+e).
__device__ __forceinline__ v16h ldfrag(const _Float16* p) {
  f16x8 lo = *(const f16x8*)p;
  f16x8 hi = *(const f16x8*)(p + 16);
  v16h r;
#pragma unroll
  for (int j = 0; j < 8; ++j) { r[j] = lo[j]; r[j + 8] = hi[j]; }
  return r;
}

// ---------------------------------------------------------------------------
// Elementwise converts / packs
// ---------------------------------------------------------------------------
__global__ __launch_bounds__(256)
void cvt_f32_f16(const float* __restrict__ s, _Float16* __restrict__ d, int n) {
  int i = blockIdx.x * 256 + threadIdx.x;
  if (i < n) d[i] = (_Float16)s[i];
}

// W [Kd][Nd] f32 -> WT [Nd][Kd] f16 (one-time; lets GEMM stream both operands
// as contiguous 16-byte runs straight into LDS).
__global__ __launch_bounds__(256)
void cvt_transpose_f32_f16(const float* __restrict__ s, _Float16* __restrict__ d,
                           int Kd, int Nd) {
  int i = blockIdx.x * 256 + threadIdx.x;
  if (i >= Kd * Nd) return;
  int k = i / Nd, n = i - k * Nd;
  d[(size_t)n * Kd + k] = (_Float16)s[i];
}

// V: [S][NKV*64] f32 -> VT [NKV][64][S] f16 (dim-major so the P*V operand
// tile is contiguous per dim row).
__global__ __launch_bounds__(256)
void pack_vT(const float* __restrict__ s, _Float16* __restrict__ d) {
  int i = blockIdx.x * 256 + threadIdx.x;      // < SEQ*KVDIM
  if (i >= SEQ * KVDIM) return;
  int sq = i >> 9;
  int r  = i & 511;
  int hk = r >> 6;
  int dd = r & 63;
  d[(((size_t)hk * HDIM) + dd) * SEQ + sq] = (_Float16)s[i];
}

// ---------------------------------------------------------------------------
// Tiled WMMA GEMM: C[M,N] = A[M,K](f16) * B[K,N] + bias, f32 out.
// B supplied pre-transposed: BT[N][K]. Block tile 128x128, 8 waves (2x4),
// each wave 64x32 => 8 WMMA per 32-deep k-step. LDS tiles double-buffered and
// filled with async Global->LDS copies overlapped with compute.
// ---------------------------------------------------------------------------
__global__ __launch_bounds__(256, 2)
void gemm_wmma(const _Float16* __restrict__ A, const _Float16* __restrict__ BT,
               const float* __restrict__ bias, float* __restrict__ C,
               int M, int N, int K) {
  __shared__ __align__(16) _Float16 sA [2][128 * 32];  // [row][k]  2x8 KB
  __shared__ __align__(16) _Float16 sBT[2][128 * 32];  // [col][k]  2x8 KB
  const int tid  = threadIdx.x;
  const int wave = tid >> 5, lane = tid & 31;
  const int hl   = lane >> 4, l16 = lane & 15;
  const int bm = blockIdx.y * 128, bn = blockIdx.x * 128;
  const int wm = (wave >> 2) * 64, wn = (wave & 3) * 32;

  const v8f vzero = {0.f, 0.f, 0.f, 0.f, 0.f, 0.f, 0.f, 0.f};
  v8f acc[4][2];
#pragma unroll
  for (int t = 0; t < 4; ++t)
#pragma unroll
    for (int u = 0; u < 2; ++u) acc[t][u] = vzero;

  // 4 async b128 copies per thread per tile (2 A + 2 BT).
  auto issue = [&](int buf, int k0) {
#pragma unroll
    for (int i = 0; i < 2; ++i) {
      int ch = tid + i * 256;                 // 512 chunks of 8 f16
      int r  = ch >> 2, c = (ch & 3) * 8;     // 128 rows x 32 k
      cp16(&A [(size_t)(bm + r) * K + k0 + c], &sA [buf][r * 32 + c]);
      cp16(&BT[(size_t)(bn + r) * K + k0 + c], &sBT[buf][r * 32 + c]);
    }
  };

  issue(0, 0);
  int buf = 0;
  for (int k0 = 0; k0 < K; k0 += 32) {
    const bool more = (k0 + 32) < K;
    if (more) issue(buf ^ 1, k0 + 32);        // prefetch next tile
    if (more) async_wait_le4(); else async_wait_0();
    __syncthreads();                           // current tile visible to all

    v16h bfrag[2];
#pragma unroll
    for (int u = 0; u < 2; ++u)
      bfrag[u] = ldfrag(&sBT[buf][(wn + u * 16 + l16) * 32 + hl * 8]);
#pragma unroll
    for (int t = 0; t < 4; ++t) {
      v16h afrag = ldfrag(&sA[buf][(wm + t * 16 + l16) * 32 + hl * 8]);
#pragma unroll
      for (int u = 0; u < 2; ++u)
        acc[t][u] = wmma16(afrag, bfrag[u], acc[t][u]);
    }
    __syncthreads();                           // done reading buf
    buf ^= 1;
  }

  // Epilogue: C layout => row = v + 8*(lane>=16), col = lane&15.
#pragma unroll
  for (int t = 0; t < 4; ++t)
#pragma unroll
    for (int u = 0; u < 2; ++u) {
      int col = bn + wn + u * 16 + l16;
      float bs = bias ? bias[col] : 0.0f;
#pragma unroll
      for (int v = 0; v < 8; ++v) {
        int row = bm + wm + t * 16 + v + 8 * hl;
        C[(size_t)row * N + col] = acc[t][u][v] + bs;
      }
    }
}

// ---------------------------------------------------------------------------
// Per-(token, head) layernorm + RoPE; one wave32 per 64-elem head row.
// in: [S][nheads*64] f32 ; out: [nheads][S][64] f16
// ---------------------------------------------------------------------------
__global__ __launch_bounds__(256)
void qk_prep(const float* __restrict__ in, const float* __restrict__ g,
             const float* __restrict__ b, _Float16* __restrict__ out,
             int nheads) {
  const int wave = threadIdx.x >> 5, lane = threadIdx.x & 31;
  const int idx = blockIdx.x * 8 + wave;
  const int s = idx / nheads, h = idx % nheads;
  const float* row = in + ((size_t)s * nheads + h) * HDIM;
  float x0 = row[lane], x1 = row[lane + 32];

  float sum = x0 + x1;
#pragma unroll
  for (int off = 16; off >= 1; off >>= 1) sum += __shfl_xor(sum, off, 32);
  float mu = sum * (1.0f / 64.0f);
  float d0 = x0 - mu, d1 = x1 - mu;
  float vs = d0 * d0 + d1 * d1;
#pragma unroll
  for (int off = 16; off >= 1; off >>= 1) vs += __shfl_xor(vs, off, 32);
  float inv = rsqrtf(vs * (1.0f / 64.0f) + 1e-5f);
  float y0 = d0 * inv * g[lane]      + b[lane];
  float y1 = d1 * inv * g[lane + 32] + b[lane + 32];

  // RoPE (emb = concat(freqs, freqs) => freq index = lane for both halves).
  float invf = __powf(10000.0f, -(float)lane / 32.0f);
  float ang  = (float)s * invf;
  float c = cosf(ang), sn = sinf(ang);
  float o0 = y0 * c - y1 * sn;      // rotate_half: [-x2, x1]
  float o1 = y1 * c + y0 * sn;

  _Float16* dst = out + ((size_t)h * SEQ + s) * HDIM;
  dst[lane]      = (_Float16)o0;
  dst[lane + 32] = (_Float16)o1;
}

// ---------------------------------------------------------------------------
// Flash-style GQA attention. Block = (head, 128-query tile), 8 waves,
// each wave owns 16 query rows; streams double-buffered 64-key tiles.
// Q:[32][S][64] K:[8][S][64] VT:[8][64][S] f16 ; O:[S][2048] f16
// ---------------------------------------------------------------------------
__global__ __launch_bounds__(256, 2)
void attn_fmha(const _Float16* __restrict__ Q, const _Float16* __restrict__ Kh,
               const _Float16* __restrict__ VT, _Float16* __restrict__ O) {
  __shared__ __align__(16) _Float16 sK [2][64 * 64];    // [key][dim] 2x8 KB
  __shared__ __align__(16) _Float16 sVT[2][64 * 64];    // [dim][key] 2x8 KB
  __shared__ __align__(16) _Float16 sP [8][16 * 64];    // per-wave P  16 KB
  const int tid  = threadIdx.x;
  const int wave = tid >> 5, lane = tid & 31;
  const int hl   = lane >> 4, l16 = lane & 15;
  const int h  = blockIdx.y;          // q head
  const int hk = h >> 2;              // kv head (n_rep = 4, grouped-adjacent)
  const int qw = blockIdx.x * 128 + wave * 16;

  // Q fragments (rows qw..qw+15, two 32-wide head-dim chunks), loaded once.
  v16h aq[2];
#pragma unroll
  for (int c = 0; c < 2; ++c)
    aq[c] = ldfrag(&Q[(((size_t)h * SEQ) + qw + l16) * HDIM + c * 32 + hl * 8]);

  const v8f vzero = {0.f, 0.f, 0.f, 0.f, 0.f, 0.f, 0.f, 0.f};
  float m[8], lsum[8];
  v8f oacc[4];
#pragma unroll
  for (int v = 0; v < 8; ++v) { m[v] = -3.0e38f; lsum[v] = 0.0f; }
#pragma unroll
  for (int t = 0; t < 4; ++t) oacc[t] = vzero;

  // 4 async b128 copies per thread per key tile (2 K + 2 VT).
  auto issueKV = [&](int buf, int kb) {
#pragma unroll
    for (int i = 0; i < 2; ++i) {
      int ch = tid + i * 256;                 // 512 chunks of 8 f16
      int rr = ch >> 3, cc = (ch & 7) * 8;    // 64 rows x 64 cols
      cp16(&Kh[(((size_t)hk * SEQ) + kb + rr) * HDIM + cc],
           &sK[buf][rr * 64 + cc]);
      cp16(&VT[(((size_t)hk * HDIM) + rr) * SEQ + kb + cc],
           &sVT[buf][rr * 64 + cc]);
    }
  };

  issueKV(0, 0);
  int buf = 0;
  for (int kb = 0; kb < SEQ; kb += 64) {
    const bool more = (kb + 64) < SEQ;
    if (more) issueKV(buf ^ 1, kb + 64);
    if (more) async_wait_le4(); else async_wait_0();
    __syncthreads();

    // S tile = Q * K^T  (16q x 64k per wave: 4 col-tiles x 2 k-chunks)
    v8f sacc[4];
#pragma unroll
    for (int u = 0; u < 4; ++u) {
      sacc[u] = vzero;
#pragma unroll
      for (int c = 0; c < 2; ++c) {
        v16h bk = ldfrag(&sK[buf][(u * 16 + l16) * 64 + c * 32 + hl * 8]);
        sacc[u] = wmma16(aq[c], bk, sacc[u]);
      }
    }

    // Online softmax: rows live as (v + 8*hl); cols across each 16-lane half.
    float mnew[8], alpha[8];
#pragma unroll
    for (int v = 0; v < 8; ++v) {
      float tm = -3.0e38f;
#pragma unroll
      for (int u = 0; u < 4; ++u) {
        sacc[u][v] *= 0.125f;                  // 1/sqrt(64)
        tm = fmaxf(tm, sacc[u][v]);
      }
#pragma unroll
      for (int off = 8; off >= 1; off >>= 1)
        tm = fmaxf(tm, __shfl_xor(tm, off, 32));
      mnew[v]  = fmaxf(m[v], tm);
      alpha[v] = __expf(m[v] - mnew[v]);
      m[v] = mnew[v];
      float rs = 0.0f;
#pragma unroll
      for (int u = 0; u < 4; ++u) {
        float p = __expf(sacc[u][v] - mnew[v]);
        rs += p;
        sP[wave][(v + 8 * hl) * 64 + u * 16 + l16] = (_Float16)p;
      }
#pragma unroll
      for (int off = 8; off >= 1; off >>= 1) rs += __shfl_xor(rs, off, 32);
      lsum[v] = lsum[v] * alpha[v] + rs;
#pragma unroll
      for (int t = 0; t < 4; ++t) oacc[t][v] *= alpha[v];
    }

    // O += P * V  (P re-read from per-wave LDS in the WMMA A layout)
#pragma unroll
    for (int c = 0; c < 2; ++c) {
      v16h ap = ldfrag(&sP[wave][l16 * 64 + c * 32 + hl * 8]);
#pragma unroll
      for (int t = 0; t < 4; ++t) {
        v16h bv = ldfrag(&sVT[buf][(t * 16 + l16) * 64 + c * 32 + hl * 8]);
        oacc[t] = wmma16(ap, bv, oacc[t]);
      }
    }
    __syncthreads();
    buf ^= 1;
  }

  // Normalize and pack into [seq][heads*64] f16 for the output projection.
#pragma unroll
  for (int t = 0; t < 4; ++t)
#pragma unroll
    for (int v = 0; v < 8; ++v) {
      int row = qw + v + 8 * hl;
      O[(size_t)row * HIDDEN + h * HDIM + t * 16 + l16] =
          (_Float16)(oacc[t][v] / lsum[v]);
    }
}

// ---------------------------------------------------------------------------
// Host orchestration
// ---------------------------------------------------------------------------
extern "C" void kernel_launch(void* const* d_in, const int* in_sizes, int n_in,
                              void* d_out, int out_size, void* d_ws,
                              size_t ws_size, hipStream_t stream) {
  (void)in_sizes; (void)n_in; (void)out_size; (void)ws_size;
  const float* x   = (const float*)d_in[0];
  const float* Wq  = (const float*)d_in[1];
  const float* bq  = (const float*)d_in[2];
  const float* Wk  = (const float*)d_in[3];
  const float* bk  = (const float*)d_in[4];
  const float* Wv  = (const float*)d_in[5];
  const float* bv  = (const float*)d_in[6];
  const float* Wo  = (const float*)d_in[7];
  const float* bo  = (const float*)d_in[8];
  const float* qg  = (const float*)d_in[9];
  const float* qb  = (const float*)d_in[10];
  const float* kg  = (const float*)d_in[11];
  const float* kbb = (const float*)d_in[12];

  char* ws = (char*)d_ws;
  size_t off = 0;
  auto alloc = [&](size_t bytes) {
    void* p = ws + off;
    off += (bytes + 255) & ~(size_t)255;
    return p;
  };
  _Float16* x16   = (_Float16*)alloc((size_t)SEQ * HIDDEN * 2);
  _Float16* WqT16 = (_Float16*)alloc((size_t)HIDDEN * HIDDEN * 2);
  _Float16* WkT16 = (_Float16*)alloc((size_t)HIDDEN * KVDIM * 2);
  _Float16* WvT16 = (_Float16*)alloc((size_t)HIDDEN * KVDIM * 2);
  _Float16* WoT16 = (_Float16*)alloc((size_t)HIDDEN * HIDDEN * 2);
  float*    Q32   = (float*)   alloc((size_t)SEQ * HIDDEN * 4);
  float*    K32   = (float*)   alloc((size_t)SEQ * KVDIM * 4);
  float*    V32   = (float*)   alloc((size_t)SEQ * KVDIM * 4);
  _Float16* Q16   = (_Float16*)alloc((size_t)NHEADS * SEQ * HDIM * 2);
  _Float16* K16   = (_Float16*)alloc((size_t)NKV * SEQ * HDIM * 2);
  _Float16* VT16  = (_Float16*)alloc((size_t)NKV * HDIM * SEQ * 2);
  _Float16* O16   = (_Float16*)alloc((size_t)SEQ * HIDDEN * 2);

  cvt_f32_f16<<<(SEQ * HIDDEN + 255) / 256, 256, 0, stream>>>(
      x, x16, SEQ * HIDDEN);
  cvt_transpose_f32_f16<<<(HIDDEN * HIDDEN + 255) / 256, 256, 0, stream>>>(
      Wq, WqT16, HIDDEN, HIDDEN);
  cvt_transpose_f32_f16<<<(HIDDEN * KVDIM + 255) / 256, 256, 0, stream>>>(
      Wk, WkT16, HIDDEN, KVDIM);
  cvt_transpose_f32_f16<<<(HIDDEN * KVDIM + 255) / 256, 256, 0, stream>>>(
      Wv, WvT16, HIDDEN, KVDIM);
  cvt_transpose_f32_f16<<<(HIDDEN * HIDDEN + 255) / 256, 256, 0, stream>>>(
      Wo, WoT16, HIDDEN, HIDDEN);

  // Projections (f16 WMMA, f32 accum + bias)
  gemm_wmma<<<dim3(HIDDEN / 128, SEQ / 128), 256, 0, stream>>>(
      x16, WqT16, bq, Q32, SEQ, HIDDEN, HIDDEN);
  gemm_wmma<<<dim3(KVDIM / 128, SEQ / 128), 256, 0, stream>>>(
      x16, WkT16, bk, K32, SEQ, KVDIM, HIDDEN);
  gemm_wmma<<<dim3(KVDIM / 128, SEQ / 128), 256, 0, stream>>>(
      x16, WvT16, bv, V32, SEQ, KVDIM, HIDDEN);

  // Per-head QK layernorm + RoPE, pack head-major f16; V packed dim-major.
  qk_prep<<<(SEQ * NHEADS) / 8, 256, 0, stream>>>(Q32, qg, qb, Q16, NHEADS);
  qk_prep<<<(SEQ * NKV) / 8, 256, 0, stream>>>(K32, kg, kbb, K16, NKV);
  pack_vT<<<(SEQ * KVDIM + 255) / 256, 256, 0, stream>>>(V32, VT16);

  // GQA flash attention
  attn_fmha<<<dim3(SEQ / 128, NHEADS), 256, 0, stream>>>(Q16, K16, VT16, O16);

  // Output projection -> f32 result
  gemm_wmma<<<dim3(HIDDEN / 128, SEQ / 128), 256, 0, stream>>>(
      O16, WoT16, bo, (float*)d_out, SEQ, HIDDEN, HIDDEN);
}
